// GPT_RWKV_78391743086622
// MI455X (gfx1250) — compile-verified, matrix operand order checked by reading, not run on hardware
//
#include <hip/hip_runtime.h>
#include <hip/hip_bf16.h>

#define BATCH 4
#define TLEN  1024
#define CDIM  512
#define HEADS 8
#define HDIM  64
#define MTOK  4096   // BATCH*TLEN
#define LAYERS 8
#define FFDIM 2048

typedef __attribute__((ext_vector_type(16))) __bf16 v16bf;
typedef __attribute__((ext_vector_type(8)))  float  v8f;

#if __has_builtin(__builtin_amdgcn_global_load_async_to_lds_b128)
#define USE_ASYNC_STAGE 1
typedef int v4i_vs __attribute__((vector_size(16)));
typedef __attribute__((address_space(1))) v4i_vs* g_v4i_p;   // global v4i*
typedef __attribute__((address_space(3))) v4i_vs* l_v4i_p;   // LDS v4i*
#define ASYNC_LD_B128(gsrc, ldst) \
    __builtin_amdgcn_global_load_async_to_lds_b128( \
        (g_v4i_p)(uintptr_t)(gsrc), (l_v4i_p)(uintptr_t)(ldst), 0, 0)
#else
#define USE_ASYNC_STAGE 0
#endif
#if __has_builtin(__builtin_amdgcn_s_wait_asynccnt)
#define WAIT_ASYNC() __builtin_amdgcn_s_wait_asynccnt(0)
#else
#define WAIT_ASYNC() asm volatile("s_wait_asynccnt 0x0" ::: "memory")
#endif

__device__ __forceinline__ unsigned short f2bf(float f) {
    unsigned int u = __float_as_uint(f);
    unsigned int r = (u + 0x7FFFu + ((u >> 16) & 1u)) >> 16;
    return (unsigned short)r;
}
__device__ __forceinline__ float sigm(float x) { return 1.f / (1.f + __expf(-x)); }

// ------------------------------------------------ weight convert + transpose
// in: [L,Kd,Nd] f32 row-major  ->  out: [L,Nd,Kd] bf16 row-major
__global__ void w2bfT_kernel(const float* __restrict__ s, unsigned short* __restrict__ d,
                             int Kd, int Nd, int total)
{
    int i = blockIdx.x * 256 + threadIdx.x;
    if (i >= total) return;
    int kn = Kd * Nd;
    int l = i / kn, rem = i - l * kn;
    int k = rem / Nd, n = rem - k * Nd;
    d[(size_t)l * kn + (size_t)n * Kd + k] = f2bf(s[i]);
}

// ------------------------------------------------ fused WMMA GEMM
// D[M,N] = Amix[M,K] @ B[K,N]; B given pre-transposed as Bt[N,K] (bf16).
// A source: bf16 buffer, or f32 H with fused token-shift mix -> bf16.
// Tile BM=64, BN=64, BK=32; 256 threads = 8 waves; 2 accumulators/wave.
enum { ASRC_BF16 = 0, ASRC_MIXF32 = 1 };
enum { EPI_STORE = 0, EPI_ADD = 1, EPI_BF16 = 2, EPI_TANH_BF16 = 3, EPI_SIG_BF16 = 4,
       EPI_RELU2_BF16 = 5, EPI_DECAY = 6, EPI_SIGBIAS = 7, EPI_VBLEND = 8 };

struct BFrag { uint4 lo, hi; };

__global__ __launch_bounds__(256)
void gemm_bf16(const void* __restrict__ Asrc, const float* __restrict__ mv,
               const unsigned short* __restrict__ Bt,
               float* __restrict__ D, unsigned short* __restrict__ Dbf,
               const float* __restrict__ aux0, const float* __restrict__ aux1,
               int Kdim, int Ndim, int modeA, int modeE)
{
    __shared__ __align__(16) unsigned short As[64 * 32];   // [m][k]
    __shared__ __align__(16) unsigned short Bs[64 * 32];   // [n][k]

    const int tid  = threadIdx.x;
    const int bm   = blockIdx.x * 64;
    const int bn   = blockIdx.y * 64;
    const int wave = tid >> 5, lane = tid & 31;
    const int wn   = (wave & 3) * 16;    // 4 n-subtiles
    const int wmg  = (wave >> 2) * 16;   // acc0 rows [wmg,16), acc1 rows [wmg+32,16)

    v8f acc0 = {0.f,0.f,0.f,0.f,0.f,0.f,0.f,0.f};
    v8f acc1 = {0.f,0.f,0.f,0.f,0.f,0.f,0.f,0.f};

    const int lm   = lane & 15;
    const int kgrp = (lane >> 4) * 8;            // ISA 16-bit A/B fragment layout
    const int sr   = tid >> 2;                   // staging row 0..63
    const int scg  = (tid & 3) * 8;              // staging col group (k)
    const bool bfull = (bn + sr < Ndim);         // only false for the single N=32 GEMM

    for (int k0 = 0; k0 < Kdim; k0 += 32) {
        // ---- stage A tile 64x32 ----
        if (modeA == ASRC_BF16) {
            const unsigned short* src =
                (const unsigned short*)Asrc + (size_t)(bm + sr) * Kdim + k0 + scg;
#if USE_ASYNC_STAGE
            ASYNC_LD_B128(src, &As[sr * 32 + scg]);
#else
            *(uint4*)&As[sr * 32 + scg] = *(const uint4*)src;
#endif
        } else {
            int tok = bm + sr;
            const float* hc = (const float*)Asrc + (size_t)tok * CDIM + k0 + scg;
            float4 h0 = *(const float4*)hc;
            float4 h1 = *(const float4*)(hc + 4);
            float4 p0 = make_float4(0.f, 0.f, 0.f, 0.f), p1 = p0;
            if ((tok & (TLEN - 1)) != 0) {
                p0 = *(const float4*)(hc - CDIM);
                p1 = *(const float4*)(hc - CDIM + 4);
            }
            float4 m0 = *(const float4*)(mv + k0 + scg);
            float4 m1 = *(const float4*)(mv + k0 + scg + 4);
            uint4 pk;
            pk.x = (unsigned)f2bf(h0.x + (p0.x - h0.x) * m0.x)
                 | ((unsigned)f2bf(h0.y + (p0.y - h0.y) * m0.y) << 16);
            pk.y = (unsigned)f2bf(h0.z + (p0.z - h0.z) * m0.z)
                 | ((unsigned)f2bf(h0.w + (p0.w - h0.w) * m0.w) << 16);
            pk.z = (unsigned)f2bf(h1.x + (p1.x - h1.x) * m1.x)
                 | ((unsigned)f2bf(h1.y + (p1.y - h1.y) * m1.y) << 16);
            pk.w = (unsigned)f2bf(h1.z + (p1.z - h1.z) * m1.z)
                 | ((unsigned)f2bf(h1.w + (p1.w - h1.w) * m1.w) << 16);
            *(uint4*)&As[sr * 32 + scg] = pk;
        }
        // ---- stage B tile 64(n)x32(k), already transposed in memory ----
        if (bfull) {
            const unsigned short* srcB = Bt + (size_t)(bn + sr) * Kdim + k0 + scg;
#if USE_ASYNC_STAGE
            ASYNC_LD_B128(srcB, &Bs[sr * 32 + scg]);
#else
            *(uint4*)&Bs[sr * 32 + scg] = *(const uint4*)srcB;
#endif
        } else {
            *(uint4*)&Bs[sr * 32 + scg] = make_uint4(0u, 0u, 0u, 0u);
        }
        // prefetch next weight k-tile (L2 resident; emits global_prefetch_b8)
        if (k0 + 32 < Kdim && bfull)
            __builtin_prefetch(Bt + (size_t)(bn + sr) * Kdim + k0 + 32 + scg, 0, 1);

#if USE_ASYNC_STAGE
        WAIT_ASYNC();
#endif
        __syncthreads();

        BFrag fa0, fa1, fb;
        {
            const unsigned short* ap0 = &As[(wmg + lm) * 32 + kgrp];
            fa0.lo = *(const uint4*)ap0;           fa0.hi = *(const uint4*)(ap0 + 16);
            const unsigned short* ap1 = &As[(wmg + 32 + lm) * 32 + kgrp];
            fa1.lo = *(const uint4*)ap1;           fa1.hi = *(const uint4*)(ap1 + 16);
            const unsigned short* bp = &Bs[(wn + lm) * 32 + kgrp];
            fb.lo = *(const uint4*)bp;             fb.hi = *(const uint4*)(bp + 16);
        }
        v16bf av0 = __builtin_bit_cast(v16bf, fa0);
        v16bf av1 = __builtin_bit_cast(v16bf, fa1);
        v16bf bv  = __builtin_bit_cast(v16bf, fb);
        acc0 = __builtin_amdgcn_wmma_f32_16x16x32_bf16(false, av0, false, bv,
                                                       (short)0, acc0, false, false);
        acc1 = __builtin_amdgcn_wmma_f32_16x16x32_bf16(false, av1, false, bv,
                                                       (short)0, acc1, false, false);
        __syncthreads();
    }

    // ---- epilogue: VGPR v -> M = v + 8*(lane>=16), N = lane&15 ----
    const int gn = bn + wn + lm;
    if (gn < Ndim) {
        const int gmA = bm + wmg + ((lane >> 4) * 8);
        #pragma unroll
        for (int half = 0; half < 2; ++half) {
            v8f acc = half ? acc1 : acc0;
            int gmb = gmA + half * 32;
            #pragma unroll
            for (int v = 0; v < 8; ++v) {
                size_t idx = (size_t)(gmb + v) * Ndim + gn;
                float val = acc[v];
                switch (modeE) {
                    case EPI_STORE:      D[idx] = val; break;
                    case EPI_ADD:        D[idx] += val; break;
                    case EPI_BF16:       Dbf[idx] = f2bf(val); break;
                    case EPI_TANH_BF16:  Dbf[idx] = f2bf(tanhf(val)); break;
                    case EPI_SIG_BF16:   Dbf[idx] = f2bf(sigm(val)); break;
                    case EPI_RELU2_BF16: { float r = val > 0.f ? val : 0.f;
                                           Dbf[idx] = f2bf(r * r); } break;
                    case EPI_DECAY: { float u = aux0[gn] + val;
                                      float w = -log1pf(__expf(-u)) - 0.5f;
                                      D[idx] = __expf(-__expf(w)); } break;
                    case EPI_SIGBIAS:    D[idx] = sigm(aux0[gn] + val); break;
                    case EPI_VBLEND: { float s = sigm(aux0[gn] + val);
                                       float vv = D[idx];
                                       D[idx] = vv + (aux1[idx] - vv) * s; } break;
                }
            }
        }
    }
}

// ------------------------------------------------ LayerNorm (per token, C=512)
__global__ __launch_bounds__(256)
void layernorm_kernel(const float* __restrict__ X, const float* __restrict__ w,
                      const float* __restrict__ b, float* __restrict__ H)
{
    int tok = blockIdx.x, tid = threadIdx.x;
    int wave = tid >> 5, lane = tid & 31;
    __shared__ float red[8];
    __shared__ float stat[2];
    size_t base = (size_t)tok * CDIM;
    float x0 = X[base + tid], x1 = X[base + tid + 256];
    float s = x0 + x1;
    #pragma unroll
    for (int d = 16; d; d >>= 1) s += __shfl_xor(s, d, 32);
    if (lane == 0) red[wave] = s;
    __syncthreads();
    if (tid == 0) { float t = 0.f; for (int i = 0; i < 8; ++i) t += red[i]; stat[0] = t * (1.f / 512.f); }
    __syncthreads();
    float mu = stat[0];
    float d0 = x0 - mu, d1 = x1 - mu;
    float vs = d0 * d0 + d1 * d1;
    #pragma unroll
    for (int d = 16; d; d >>= 1) vs += __shfl_xor(vs, d, 32);
    if (lane == 0) red[wave] = vs;
    __syncthreads();
    if (tid == 0) { float t = 0.f; for (int i = 0; i < 8; ++i) t += red[i]; stat[1] = rsqrtf(t * (1.f / 512.f) + 1e-5f); }
    __syncthreads();
    float inv = stat[1];
    H[base + tid]       = d0 * inv * w[tid]       + b[tid];
    H[base + tid + 256] = d1 * inv * w[tid + 256] + b[tid + 256];
}

// ------------------------------------------------ kk normalize + k update
__global__ __launch_bounds__(256)
void kk_kernel(float* __restrict__ K, const float* __restrict__ A,
               const float* __restrict__ kkp_, const float* __restrict__ kap_,
               float* __restrict__ KK)
{
    int tok = blockIdx.x;
    int wave = threadIdx.x >> 5, lane = threadIdx.x & 31;  // wave == head
    int c0 = wave * 64 + lane, c1 = c0 + 32;
    size_t base = (size_t)tok * CDIM;
    float k0 = K[base + c0], k1 = K[base + c1];
    float q0 = k0 * kkp_[c0], q1 = k1 * kkp_[c1];
    float ss = q0 * q0 + q1 * q1;
    #pragma unroll
    for (int d = 16; d; d >>= 1) ss += __shfl_xor(ss, d, 32);
    float inv = 1.f / fmaxf(sqrtf(ss), 1e-12f);
    KK[base + c0] = q0 * inv;
    KK[base + c1] = q1 * inv;
    K[base + c0] = k0 * (1.f + (A[base + c0] - 1.f) * kap_[c0]);
    K[base + c1] = k1 * (1.f + (A[base + c1] - 1.f) * kap_[c1]);
}

// ------------------------------------------------ RWKV7 delta-rule scan
__global__ __launch_bounds__(64)
void scan_kernel(const float* __restrict__ R, const float* __restrict__ DEC,
                 const float* __restrict__ K, const float* __restrict__ V,
                 const float* __restrict__ KK, const float* __restrict__ A,
                 float* __restrict__ O)
{
    int bh = blockIdx.x;
    int b = bh >> 3, h = bh & 7;
    int i = threadIdx.x;
    __shared__ float lr[64], ld[64], lk[64], lv[64], lkk[64], la[64];
    float S[64];
    #pragma unroll
    for (int j = 0; j < 64; ++j) S[j] = 0.f;
    size_t base = ((size_t)b * TLEN) * CDIM + h * HDIM;
    for (int t = 0; t < TLEN; ++t) {
        size_t off = base + (size_t)t * CDIM;
        lr[i] = R[off + i];  ld[i] = DEC[off + i]; lk[i] = K[off + i];
        lv[i] = V[off + i];  lkk[i] = KK[off + i]; la[i] = A[off + i];
        __syncthreads();
        float vi = lv[i];
        float sa = 0.f;                 // S . a_t, a_t = -kk
        #pragma unroll
        for (int j = 0; j < 64; ++j) sa -= S[j] * lkk[j];
        float o = 0.f;
        #pragma unroll
        for (int j = 0; j < 64; ++j) {
            float bj = lkk[j] * la[j];  // b_t = kk * a
            S[j] = S[j] * ld[j] + sa * bj + vi * lk[j];
            o += S[j] * lr[j];
        }
        O[off + i] = o;
        __syncthreads();
    }
}

// ------------------------------------------------ GroupNorm + rk bonus + gate -> bf16
__global__ __launch_bounds__(256)
void gnorm_gate_kernel(const float* __restrict__ O, const float* __restrict__ R,
                       const float* __restrict__ K, const float* __restrict__ V,
                       const float* __restrict__ G, const float* __restrict__ lnxw,
                       const float* __restrict__ lnxb, const float* __restrict__ rkp,
                       unsigned short* __restrict__ out)
{
    int tok = blockIdx.x;
    int wave = threadIdx.x >> 5, lane = threadIdx.x & 31;   // wave == head
    int c0 = wave * 64 + lane, c1 = c0 + 32;
    size_t base = (size_t)tok * CDIM;
    float o0 = O[base + c0], o1 = O[base + c1];
    float s = o0 + o1;
    #pragma unroll
    for (int d = 16; d; d >>= 1) s += __shfl_xor(s, d, 32);
    float mu = s * (1.f / 64.f);
    float d0 = o0 - mu, d1 = o1 - mu;
    float vs = d0 * d0 + d1 * d1;
    #pragma unroll
    for (int d = 16; d; d >>= 1) vs += __shfl_xor(vs, d, 32);
    float inv = rsqrtf(vs * (1.f / 64.f) + 64e-5f);
    float p = R[base + c0] * K[base + c0] * rkp[c0]
            + R[base + c1] * K[base + c1] * rkp[c1];
    #pragma unroll
    for (int d = 16; d; d >>= 1) p += __shfl_xor(p, d, 32);
    float out0 = (d0 * inv * lnxw[c0] + lnxb[c0] + p * V[base + c0]) * G[base + c0];
    float out1 = (d1 * inv * lnxw[c1] + lnxb[c1] + p * V[base + c1]) * G[base + c1];
    out[base + c0] = f2bf(out0);
    out[base + c1] = f2bf(out1);
}

// ------------------------------------------------ img -> [B,T,C] (+pos)
__global__ void prep_kernel(const float* __restrict__ img, const float* __restrict__ pe,
                            float* __restrict__ X)
{
    int i = blockIdx.x * 256 + threadIdx.x;   // B*T*C = 2^21
    int c = i & 511;
    int t = (i >> 9) & 1023;
    int b = i >> 19;
    int s = t >> 7, ppix = t & 127;
    int y = ppix >> 4, xw = ppix & 15;
    int eb = b * 8 + s;
    X[i] = img[(((size_t)eb * 512 + c) * 8 + y) * 16 + xw] + pe[ppix * 512 + c];
}

__global__ void writeout_kernel(const float* __restrict__ Hf, float* __restrict__ out)
{
    int i = blockIdx.x * 256 + threadIdx.x;   // EB*C*8*16 = 2^21
    int xw = i & 15, y = (i >> 4) & 7, c = (i >> 7) & 511, eb = i >> 16;
    int b = eb >> 3, s = eb & 7;
    int t = s * 128 + y * 16 + xw;
    out[i] = Hf[((size_t)(b * TLEN + t)) * CDIM + c];
}

// ================================================================ host
extern "C" void kernel_launch(void* const* d_in, const int* in_sizes, int n_in,
                              void* d_out, int out_size, void* d_ws, size_t ws_size,
                              hipStream_t stream)
{
    (void)in_sizes; (void)n_in; (void)out_size; (void)ws_size;

    const float* image = (const float*)d_in[0];
    const float* pos   = (const float*)d_in[1];
    const float* xm    = (const float*)d_in[2];
    const float* w0    = (const float*)d_in[3];
    const float* w1    = (const float*)d_in[4];
    const float* w2    = (const float*)d_in[5];
    const float* a0    = (const float*)d_in[6];
    const float* a1    = (const float*)d_in[7];
    const float* a2    = (const float*)d_in[8];
    const float* v0    = (const float*)d_in[9];
    const float* v1    = (const float*)d_in[10];
    const float* v2    = (const float*)d_in[11];
    const float* g1    = (const float*)d_in[12];
    const float* g2    = (const float*)d_in[13];
    const float* kkp   = (const float*)d_in[14];
    const float* kap   = (const float*)d_in[15];
    const float* rkp   = (const float*)d_in[16];
    const float* Wr    = (const float*)d_in[17];
    const float* Wk    = (const float*)d_in[18];
    const float* Wv    = (const float*)d_in[19];
    const float* Wo    = (const float*)d_in[20];
    const float* lnxw  = (const float*)d_in[21];
    const float* lnxb  = (const float*)d_in[22];
    const float* ln1w  = (const float*)d_in[23];
    const float* ln1b  = (const float*)d_in[24];
    const float* ln2w  = (const float*)d_in[25];
    const float* ln2b  = (const float*)d_in[26];
    const float* fxk   = (const float*)d_in[27];
    const float* Wkey  = (const float*)d_in[28];
    const float* Wval  = (const float*)d_in[29];
    const float* lnfw  = (const float*)d_in[30];
    const float* lnfb  = (const float*)d_in[31];

    char* p = (char*)d_ws;
    auto alloc = [&](size_t bytes) -> void* {
        void* r = (void*)p; p += (bytes + 255) & ~(size_t)255; return r;
    };
    const size_t CC = (size_t)CDIM * CDIM;
    unsigned short* bWr   = (unsigned short*)alloc(LAYERS * CC * 2);
    unsigned short* bWk   = (unsigned short*)alloc(LAYERS * CC * 2);
    unsigned short* bWv   = (unsigned short*)alloc(LAYERS * CC * 2);
    unsigned short* bWo   = (unsigned short*)alloc(LAYERS * CC * 2);
    unsigned short* bw1   = (unsigned short*)alloc((size_t)LAYERS * CDIM * 64 * 2);
    unsigned short* bw2   = (unsigned short*)alloc((size_t)LAYERS * 64 * CDIM * 2);
    unsigned short* ba1   = (unsigned short*)alloc((size_t)LAYERS * CDIM * 64 * 2);
    unsigned short* ba2   = (unsigned short*)alloc((size_t)LAYERS * 64 * CDIM * 2);
    unsigned short* bv1   = (unsigned short*)alloc((size_t)LAYERS * CDIM * 32 * 2);
    unsigned short* bv2   = (unsigned short*)alloc((size_t)LAYERS * 32 * CDIM * 2);
    unsigned short* bg1   = (unsigned short*)alloc((size_t)LAYERS * CDIM * 128 * 2);
    unsigned short* bg2   = (unsigned short*)alloc((size_t)LAYERS * 128 * CDIM * 2);
    unsigned short* bWkey = (unsigned short*)alloc((size_t)LAYERS * CDIM * FFDIM * 2);
    unsigned short* bWval = (unsigned short*)alloc((size_t)LAYERS * FFDIM * CDIM * 2);

    const size_t TC = (size_t)MTOK * CDIM;
    float* X    = (float*)alloc(TC * 4);
    float* Hb   = (float*)alloc(TC * 4);
    float* Rb   = (float*)alloc(TC * 4);
    float* Kb   = (float*)alloc(TC * 4);
    float* Vb   = (float*)alloc(TC * 4);
    float* DEC  = (float*)alloc(TC * 4);
    float* Ab   = (float*)alloc(TC * 4);
    float* Gb   = (float*)alloc(TC * 4);
    float* KKb  = (float*)alloc(TC * 4);
    float* VF   = (float*)alloc(TC * 4);
    float* Ob   = (float*)alloc(TC * 4);
    unsigned short* MIX = (unsigned short*)alloc(TC * 2);
    unsigned short* TB  = (unsigned short*)alloc((size_t)MTOK * 128 * 2);
    unsigned short* FFB = (unsigned short*)alloc((size_t)MTOK * FFDIM * 2);

    auto convT = [&](const float* s, unsigned short* d, int Kd, int Nd) {
        int total = LAYERS * Kd * Nd;
        w2bfT_kernel<<<(total + 255) / 256, 256, 0, stream>>>(s, d, Kd, Nd, total);
    };
    convT(Wr, bWr, CDIM, CDIM);   convT(Wk, bWk, CDIM, CDIM);
    convT(Wv, bWv, CDIM, CDIM);   convT(Wo, bWo, CDIM, CDIM);
    convT(w1, bw1, CDIM, 64);     convT(w2, bw2, 64, CDIM);
    convT(a1, ba1, CDIM, 64);     convT(a2, ba2, 64, CDIM);
    convT(v1, bv1, CDIM, 32);     convT(v2, bv2, 32, CDIM);
    convT(g1, bg1, CDIM, 128);    convT(g2, bg2, 128, CDIM);
    convT(Wkey, bWkey, CDIM, FFDIM);
    convT(Wval, bWval, FFDIM, CDIM);

    auto gemm = [&](const void* Am, const float* mvv, const unsigned short* Bm,
                    float* Dm, unsigned short* Dbm, const float* aux0, const float* aux1,
                    int Kd, int Nd, int modeA, int modeE) {
        dim3 grid(MTOK / 64, (Nd + 63) / 64);
        gemm_bf16<<<grid, 256, 0, stream>>>(Am, mvv, Bm, Dm, Dbm, aux0, aux1,
                                            Kd, Nd, modeA, modeE);
    };

    prep_kernel<<<(MTOK * CDIM) / 256, 256, 0, stream>>>(image, pos, X);

    for (int l = 0; l < LAYERS; ++l) {
        const float* xml = xm + (size_t)l * 6 * CDIM;

        // ---- time mix ----
        layernorm_kernel<<<MTOK, 256, 0, stream>>>(X, ln1w + l * CDIM, ln1b + l * CDIM, Hb);

        // r = mix0(h) @ Wr
        gemm(Hb, xml + 0 * CDIM, bWr + l * CC, Rb, nullptr, nullptr, nullptr,
             CDIM, CDIM, ASRC_MIXF32, EPI_STORE);
        // decay: tanh(mix1 @ w1) @ w2 -> fused softplus/exp epilogue
        gemm(Hb, xml + 1 * CDIM, bw1 + (size_t)l * CDIM * 64, nullptr, TB, nullptr, nullptr,
             CDIM, 64, ASRC_MIXF32, EPI_TANH_BF16);
        gemm(TB, nullptr, bw2 + (size_t)l * 64 * CDIM, DEC, nullptr, w0 + l * CDIM, nullptr,
             64, CDIM, ASRC_BF16, EPI_DECAY);
        // k = mix2(h) @ Wk
        gemm(Hb, xml + 2 * CDIM, bWk + l * CC, Kb, nullptr, nullptr, nullptr,
             CDIM, CDIM, ASRC_MIXF32, EPI_STORE);
        // v = mix3(h) @ Wv (+ first-layer value residual blend)
        gemm(Hb, xml + 3 * CDIM, bWv + l * CC, Vb, nullptr, nullptr, nullptr,
             CDIM, CDIM, ASRC_MIXF32, EPI_STORE);
        if (l == 0) {
            (void)hipMemcpyAsync(VF, Vb, TC * 4, hipMemcpyDeviceToDevice, stream);
        } else {
            gemm(Hb, xml + 3 * CDIM, bv1 + (size_t)l * CDIM * 32, nullptr, TB, nullptr, nullptr,
                 CDIM, 32, ASRC_MIXF32, EPI_BF16);
            gemm(TB, nullptr, bv2 + (size_t)l * 32 * CDIM, Vb, nullptr,
                 v0 + l * CDIM, VF, 32, CDIM, ASRC_BF16, EPI_VBLEND);
        }
        // a = sigmoid(a0 + (mix4 @ a1) @ a2)
        gemm(Hb, xml + 4 * CDIM, ba1 + (size_t)l * CDIM * 64, nullptr, TB, nullptr, nullptr,
             CDIM, 64, ASRC_MIXF32, EPI_BF16);
        gemm(TB, nullptr, ba2 + (size_t)l * 64 * CDIM, Ab, nullptr, a0 + l * CDIM, nullptr,
             64, CDIM, ASRC_BF16, EPI_SIGBIAS);
        // g = sigmoid(mix5 @ g1) @ g2
        gemm(Hb, xml + 5 * CDIM, bg1 + (size_t)l * CDIM * 128, nullptr, TB, nullptr, nullptr,
             CDIM, 128, ASRC_MIXF32, EPI_SIG_BF16);
        gemm(TB, nullptr, bg2 + (size_t)l * 128 * CDIM, Gb, nullptr, nullptr, nullptr,
             128, CDIM, ASRC_BF16, EPI_STORE);

        // kk normalize + in-context-lr k update
        kk_kernel<<<MTOK, 256, 0, stream>>>(Kb, Ab, kkp + l * CDIM, kap + l * CDIM, KKb);
        // sequential delta-rule scan
        scan_kernel<<<BATCH * HEADS, 64, 0, stream>>>(Rb, DEC, Kb, Vb, KKb, Ab, Ob);
        // groupnorm + rk bonus + gate -> bf16 operand for Wo
        gnorm_gate_kernel<<<MTOK, 256, 0, stream>>>(Ob, Rb, Kb, Vb, Gb,
                                                    lnxw + l * CDIM, lnxb + l * CDIM,
                                                    rkp + l * CDIM, MIX);
        // x += (o*g) @ Wo
        gemm(MIX, nullptr, bWo + l * CC, X, nullptr, nullptr, nullptr,
             CDIM, CDIM, ASRC_BF16, EPI_ADD);

        // ---- channel mix ----
        layernorm_kernel<<<MTOK, 256, 0, stream>>>(X, ln2w + l * CDIM, ln2b + l * CDIM, Hb);
        gemm(Hb, fxk + l * CDIM, bWkey + (size_t)l * CDIM * FFDIM, nullptr, FFB,
             nullptr, nullptr, CDIM, FFDIM, ASRC_MIXF32, EPI_RELU2_BF16);
        gemm(FFB, nullptr, bWval + (size_t)l * FFDIM * CDIM, X, nullptr, nullptr, nullptr,
             FFDIM, CDIM, ASRC_BF16, EPI_ADD);
    }

    layernorm_kernel<<<MTOK, 256, 0, stream>>>(X, lnfw, lnfb, Hb);
    writeout_kernel<<<(MTOK * CDIM) / 256, 256, 0, stream>>>(Hb, (float*)d_out);
}
